// VanillaRNN_27307402068315
// MI455X (gfx1250) — matrix operand order repaired and checked by reading, not compiled, not run on hardware
//
#include <hip/hip_runtime.h>

#define B_TOTAL 256
#define T_STEPS 2000
#define IN_SZ   33
#define HID     128
#define OUT_SZ  3
#define ALPHA   0.04f

typedef __attribute__((ext_vector_type(16))) __bf16 v16bf;
typedef __attribute__((ext_vector_type(8)))  __bf16 bf16x8;
typedef __attribute__((ext_vector_type(8)))  float  v8f;
typedef __attribute__((ext_vector_type(4)))  float  f32x4;
typedef __attribute__((ext_vector_type(4)))  unsigned int u32x4;
typedef __attribute__((ext_vector_type(8)))  int    i32x8;
typedef __attribute__((ext_vector_type(4)))  int    i32x4;

// Low 32 bits of a generic pointer to LDS == byte offset within the WG's LDS
// allocation (ISA 10.2).
__device__ __forceinline__ unsigned lds_off(const void* p) {
  return (unsigned)(unsigned long long)p;
}

// Issue one TDM load of a 16-row x 33-col f32 tile (row stride T*33 elems)
// from global memory into LDS at byte offset ldsByteOff. D# per ISA ch.8.
__device__ __forceinline__ void tdm_load_tile(const float* gptr, unsigned ldsByteOff) {
  unsigned long long ga = (unsigned long long)gptr;
  u32x4 g0;
  g0[0] = 1u;                                   // count=1, user descriptor
  g0[1] = ldsByteOff;                           // lds_addr (bytes)
  g0[2] = (unsigned)ga;                         // global_addr[31:0]
  g0[3] = (unsigned)((ga >> 32) & 0x01FFFFFFull) | (2u << 30); // addr hi | type=2
  i32x8 g1;
  g1[0] = 2 << 16;                              // data_size = 4 bytes
  g1[1] = IN_SZ << 16;                          // tensor_dim0 (bits 79:48, low part)
  g1[2] = 16 << 16;                             // tensor_dim1 (bits 111:80, low part)
  g1[3] = IN_SZ << 16;                          // tile_dim0 (bits 127:112)
  g1[4] = 16;                                   // tile_dim1 (bits 143:128); tile_dim2=0
  g1[5] = T_STEPS * IN_SZ;                      // tensor_dim0_stride low32
  g1[6] = 0;
  g1[7] = 0;
  i32x4 z4;
  z4[0] = 0; z4[1] = 0; z4[2] = 0; z4[3] = 0;   // groups 2/3 unused (2-D tensor)
  i32x8 z8;
  #pragma unroll
  for (int i = 0; i < 8; ++i) z8[i] = 0;
  __builtin_amdgcn_tensor_load_to_lds(g0, g1, z4, z4, z8, 0);
}

__global__ __launch_bounds__(256)
void rnn_scan_kernel(const float* __restrict__ x,
                     const float* __restrict__ W_in,
                     const float* __restrict__ b_in,
                     const float* __restrict__ W_rec,
                     float* __restrict__ hs_out)
{
  // Double-buffered bf16 hidden state (A-operand source) + double-buffered x tile.
  __shared__ __bf16 Hs[2][16][HID];
  __shared__ float  Xs[2][16 * IN_SZ];

  const int tid  = threadIdx.x;
  const int w    = tid >> 5;        // wave id 0..7, owns hidden cols [16w,16w+16)
  const int l    = tid & 31;        // lane id
  const int half = l >> 4;          // lane half (WMMA layout)
  const int nl   = l & 15;          // col (B/C layout) or row (A layout) in tile
  const int b0   = blockIdx.x * 16; // this WG's batch slice
  const int ng   = w * 16 + nl;     // global hidden column owned by this lane
  const int kb   = half * 8;        // A-fragment K base for this lane half
  const int kb2  = half * 16;       // unused; B fragments also use kb

  // ---------- one-time resident B fragments (bf16) ----------
  // Recurrence: out[m][n] = sum_k h[m][k] * W_rec[n][k]  -> B[k][n] = W_rec[n][k]
  v16bf Br[4];
  #pragma unroll
  for (int c = 0; c < 4; ++c) {
    const float* p = W_rec + (long long)ng * HID + c * 32 + kb;
    #pragma unroll
    for (int e = 0; e < 16; ++e) Br[c][e] = (__bf16)p[(e < 8) ? e : 8 + e];
  }
  // Projection: B[k][n] = W_in[n][k], K padded 33 -> 64 with zeros
  v16bf Bi0;
  {
    const float* p = W_in + (long long)ng * IN_SZ + kb;  // kb+8+15 <= 31 < 33: in range
    #pragma unroll
    for (int e = 0; e < 16; ++e) Bi0[e] = (__bf16)p[(e < 8) ? e : 8 + e];
  }
  v16bf Bi1;
  #pragma unroll
  for (int e = 0; e < 16; ++e) Bi1[e] = (__bf16)0.f;
  if (half == 0) Bi1[0] = (__bf16)W_in[(long long)ng * IN_SZ + 32];  // k==32 only
  const float bias = b_in[ng];

  // f32 master copy of this wave's 16x16 H tile, in WMMA C layout
  float hc[8];
  #pragma unroll
  for (int j = 0; j < 8; ++j) hc[j] = 0.f;

  // zero initial hidden-state buffer 0
  for (int i = tid; i < 16 * HID; i += 256)
    ((__bf16*)&Hs[0][0][0])[i] = (__bf16)0.f;

  // prefetch x tile for t = 0 via TDM
  if (w == 0) {
    tdm_load_tile(x + (long long)b0 * T_STEPS * IN_SZ, lds_off(&Xs[0][0]));
    __builtin_amdgcn_s_wait_tensorcnt(0);
  }
  __syncthreads();

  for (int t = 0; t < T_STEPS; ++t) {
    const int cur = t & 1;
    const int nxt = cur ^ 1;

    // async-prefetch next step's x tile into the other buffer
    if (w == 0 && (t + 1) < T_STEPS) {
      tdm_load_tile(x + ((long long)b0 * T_STEPS + (t + 1)) * IN_SZ,
                    lds_off(&Xs[nxt][0]));
    }

    // ---------- A fragments: x tile (row = nl), K padded to 64 ----------
    const float* xr = &Xs[cur][nl * IN_SZ];
    v16bf ax0;
    #pragma unroll
    for (int e = 0; e < 8; ++e) ax0[e]     = (__bf16)xr[kb + e];        // K kb..kb+7
    #pragma unroll
    for (int e = 0; e < 8; ++e) ax0[8 + e] = (__bf16)xr[16 + kb + e];   // K 16+kb..
    v16bf ax1;
    #pragma unroll
    for (int e = 0; e < 16; ++e) ax1[e] = (__bf16)0.f;
    if (half == 0) ax1[0] = (__bf16)xr[32];                             // k==32

    // ---------- A fragments: hidden state (contiguous b128 LDS loads) ----------
    const __bf16* hr = &Hs[cur][nl][0];
    v16bf ah[4];
    #pragma unroll
    for (int c = 0; c < 4; ++c) {
      bf16x8 lo = *(const bf16x8*)(hr + 32 * c + kb);
      bf16x8 hi = *(const bf16x8*)(hr + 32 * c + 16 + kb);
      ah[c] = __builtin_shufflevector(lo, hi,
                0,1,2,3,4,5,6,7,8,9,10,11,12,13,14,15);
    }

    // ---------- 6 chained WMMAs: bias + x*W_in^T + h*W_rec^T ----------
    v8f acc;
    #pragma unroll
    for (int j = 0; j < 8; ++j) acc[j] = bias;
    acc = __builtin_amdgcn_wmma_f32_16x16x32_bf16(false, ax0, false, Bi0,
                                                  (short)0, acc, false, false);
    acc = __builtin_amdgcn_wmma_f32_16x16x32_bf16(false, ax1, false, Bi1,
                                                  (short)0, acc, false, false);
    #pragma unroll
    for (int c = 0; c < 4; ++c)
      acc = __builtin_amdgcn_wmma_f32_16x16x32_bf16(false, ah[c], false, Br[c],
                                                    (short)0, acc, false, false);

    // ---------- leaky tanh update, publish H, stream hidden state ----------
    #pragma unroll
    for (int j = 0; j < 8; ++j) {
      const int m = half * 8 + j;                       // row in C layout
      float a  = fminf(fmaxf(acc[j], -15.f), 15.f);
      float e2 = __expf(2.f * a);
      float th = (e2 - 1.f) / (e2 + 1.f);               // tanh
      hc[j] = (1.f - ALPHA) * hc[j] + ALPHA * th;
      Hs[nxt][m][ng] = (__bf16)hc[j];
      hs_out[((long long)(b0 + m) * T_STEPS + t) * HID + ng] = hc[j];
    }

    if (w == 0 && (t + 1) < T_STEPS) __builtin_amdgcn_s_wait_tensorcnt(0);
    __syncthreads();   // next-H visible + next-x DMA complete for all waves
  }
}

__global__ __launch_bounds__(256)
void rnn_readout_kernel(const float* __restrict__ hs,
                        const float* __restrict__ W_out,
                        const float* __restrict__ b_out,
                        float* __restrict__ pred)
{
  const long long row = (long long)blockIdx.x * blockDim.x + threadIdx.x;
  if (row >= (long long)B_TOTAL * T_STEPS) return;
  const f32x4* h4 = (const f32x4*)(hs + row * HID);
  const f32x4* w0 = (const f32x4*)(W_out);
  const f32x4* w1 = (const f32x4*)(W_out + HID);
  const f32x4* w2 = (const f32x4*)(W_out + 2 * HID);
  float a0 = b_out[0], a1 = b_out[1], a2 = b_out[2];
  #pragma unroll 4
  for (int k = 0; k < HID / 4; ++k) {
    f32x4 h = h4[k], u0 = w0[k], u1 = w1[k], u2 = w2[k];
    #pragma unroll
    for (int i = 0; i < 4; ++i) {
      a0 += h[i] * u0[i];
      a1 += h[i] * u1[i];
      a2 += h[i] * u2[i];
    }
  }
  pred[row * 3 + 0] = a0;
  pred[row * 3 + 1] = a1;
  pred[row * 3 + 2] = a2;
}

extern "C" void kernel_launch(void* const* d_in, const int* in_sizes, int n_in,
                              void* d_out, int out_size, void* d_ws, size_t ws_size,
                              hipStream_t stream) {
  (void)in_sizes; (void)n_in; (void)out_size; (void)d_ws; (void)ws_size;
  const float* x     = (const float*)d_in[0];
  const float* W_in  = (const float*)d_in[1];
  const float* b_in  = (const float*)d_in[2];
  const float* W_rec = (const float*)d_in[3];
  const float* W_out = (const float*)d_in[4];
  const float* b_out = (const float*)d_in[5];

  float* pred = (float*)d_out;                                   // [B,T,3]
  float* hs   = pred + (long long)B_TOTAL * T_STEPS * OUT_SZ;    // [B,T,128]

  rnn_scan_kernel<<<B_TOTAL / 16, 256, 0, stream>>>(x, W_in, b_in, W_rec, hs);

  const int rows = B_TOTAL * T_STEPS;
  rnn_readout_kernel<<<(rows + 255) / 256, 256, 0, stream>>>(hs, W_out, b_out, pred);
}
